// BLA_51110110822841
// MI455X (gfx1250) — matrix operand order repaired and checked by reading, not compiled, not run on hardware
//
#include <hip/hip_runtime.h>
#include <hip/hip_bf16.h>
#include <stdint.h>

// ---------------------------------------------------------------------------
// CDNA5 / gfx1250: wave32, WMMA 16x16x32 bf16 with fp32 accumulation,
// TDM tensor_load_to_lds for weight staging.
// ---------------------------------------------------------------------------
typedef __attribute__((ext_vector_type(16))) __bf16 bf16x16;
typedef __attribute__((ext_vector_type(8)))  float  f32x8;
typedef __attribute__((ext_vector_type(4)))  float  f32x4;   // 16B copy unit
typedef __attribute__((ext_vector_type(4)))  unsigned int u32x4;
typedef __attribute__((ext_vector_type(8)))  int    i32x8;
typedef __attribute__((ext_vector_type(4)))  int    i32x4;

union Frag32 { bf16x16 v; f32x4 f4[2]; };                    // 32B A/B operand

__device__ __forceinline__ float sigm(float x) { return 1.0f / (1.0f + __expf(-x)); }

// ---- fp32 -> bf16 convert -------------------------------------------------
__global__ void k_cvt(const float* __restrict__ s, __bf16* __restrict__ d, long n) {
  long i = (long)blockIdx.x * blockDim.x + threadIdx.x;
  if (i < n) d[i] = (__bf16)s[i];
}

// ---- transpose convert: d[c*R + r] = s[r*C + c] ---------------------------
__global__ void k_cvtT(const float* __restrict__ s, __bf16* __restrict__ d, int R, int C) {
  long i = (long)blockIdx.x * blockDim.x + threadIdx.x;
  if (i < (long)R * C) {
    int r = (int)(i / C), c = (int)(i % C);
    d[(size_t)c * R + r] = (__bf16)s[i];
  }
}

// ---- layer-0 input projection (K=15: scalar FMA is optimal) --------------
__global__ void k_proj0(const float* __restrict__ x,
                        const float* __restrict__ wihF, const float* __restrict__ bF,
                        const float* __restrict__ wihB, const float* __restrict__ bB,
                        float* __restrict__ preA, float* __restrict__ preB) {
  long id = (long)blockIdx.x * blockDim.x + threadIdx.x;   // t*131072 + b*1024 + g
  if (id >= (long)512 * 128 * 1024) return;
  int g = (int)(id & 1023);
  int b = (int)((id >> 10) & 127);
  int t = (int)(id >> 17);
  const float* xr = x + ((size_t)b * 512 + t) * 15;
  float sf = bF[g], sb = bB[g];
  #pragma unroll
  for (int d = 0; d < 15; ++d) {
    float xv = xr[d];
    sf += xv * wihF[g * 15 + d];
    sb += xv * wihB[g * 15 + d];
  }
  preA[id] = sf;
  preB[id] = sb;
}

// ---- bf16 WMMA GEMM: out[m][n] = bias[n] + sum_k A[m][k]*W[n][k] ----------
// Software-pipelined: double-buffered A/B fragments so global loads for the
// next K-chunk are in flight while WMMAs consume the current one.
__global__ __launch_bounds__(256)
void k_gemm(const __bf16* __restrict__ A, const __bf16* __restrict__ W,
            const float* __restrict__ bias, float* __restrict__ out,
            int N, int K, int transStore) {
  const int lane = threadIdx.x & 31;
  const int wv   = threadIdx.x >> 5;
  const int half = lane >> 4;
  const int nl   = lane & 15;
  const long m0  = (long)blockIdx.y * 128 + wv * 16;
  const int  n0  = blockIdx.x * 64;

  const __bf16* arow = A + (size_t)(m0 + nl) * K;

  f32x8 acc[4] = {};
  Frag32 a[2];
  Frag32 b[2][4];

  auto load = [&](int buf, int kt) {
    a[buf].f4[0] = *(const f32x4*)(arow + kt + half * 8);
    a[buf].f4[1] = *(const f32x4*)(arow + kt + 16 + half * 8);
    #pragma unroll
    for (int nt = 0; nt < 4; ++nt) {
      const __bf16* wrow = W + (size_t)(n0 + nt * 16 + nl) * K + kt + half * 8;
      b[buf][nt].f4[0] = *(const f32x4*)(wrow);
      b[buf][nt].f4[1] = *(const f32x4*)(wrow + 16);
    }
  };
  auto mmas = [&](int buf) {
    #pragma unroll
    for (int nt = 0; nt < 4; ++nt)
      acc[nt] = __builtin_amdgcn_wmma_f32_16x16x32_bf16(
          false, a[buf].v, false, b[buf][nt].v, (short)0, acc[nt], false, false);
  };

  load(0, 0);
  for (int kt = 0; kt < K; kt += 64) {                 // K is a multiple of 64
    if (kt + 32 < K) load(1, kt + 32);
    mmas(0);
    if (kt + 64 < K) load(0, kt + 64);
    if (kt + 32 < K) mmas(1);
  }

  #pragma unroll
  for (int nt = 0; nt < 4; ++nt) {
    int n = n0 + nt * 16 + nl;
    float bv = bias[n];
    #pragma unroll
    for (int v = 0; v < 8; ++v) {
      long m = m0 + v + 8 * half;                      // D layout: row = v + 8*(lane>=16)
      float val = acc[nt][v] + bv;
      if (!transStore) out[(size_t)m * N + n] = val;
      else out[((size_t)(m >> 7) * N + n) * 128 + (m & 127)] = val;
    }
  }
}

// ---- persistent LSTM recurrence ------------------------------------------
// 8 WGs per direction, each owns 32 hidden units. Whh slice (128x256 bf16,
// 64KB) is DMA'd into LDS by the Tensor Data Mover with a single 3-D
// descriptor (256 cols x 32 rows x 4 gates, gate stride 65536 elements).
// Per step: 128x128 GEMM tile via WMMA, gate math fully in registers, h
// exchanged through L2 as bf16, device-scope monotonic atomic barrier.
__global__ __launch_bounds__(256)
void k_lstm(const float* __restrict__ preF, const float* __restrict__ preB,
            const __bf16* __restrict__ whhF, const __bf16* __restrict__ whhB,
            __bf16* __restrict__ hbuf, int hw,           // hw = 512 (bi) or 256 (uni)
            float* __restrict__ uout,                    // fp32 copy of h (top LSTM)
            unsigned* __restrict__ cnts, int T) {
  const int dir = blockIdx.x >> 3;                       // 8 WGs per direction
  const int wg  = blockIdx.x & 7;
  const int j0  = wg * 32;
  const int colbase = (hw == 512) ? dir * 256 : 0;
  const float*  pre = dir ? preB : preF;
  const __bf16* whh = dir ? whhB : whhF;
  unsigned* cnt = cnts + dir;
  const bool rev = (dir != 0);

  extern __shared__ __align__(16) char smem[];
  __bf16* Ws = (__bf16*)smem;                            // [128 rows][256 K] bf16

#if __has_builtin(__builtin_amdgcn_tensor_load_to_lds)
  if (threadIdx.x == 0) {
    // D# group0: count=1 | lds_addr | global_addr (57b) | type=2
    unsigned ldsa = (unsigned)(size_t)(void*)Ws;         // low 32b of flat = LDS offset
    uint64_t ga = (uint64_t)(size_t)(const void*)(whh + (size_t)j0 * 256);
    u32x4 g0 = { 1u, ldsa, (unsigned)ga,
                 (unsigned)((ga >> 32) & 0x01ffffffu) | (2u << 30) };
    // group1: data_size=1(2B); tensor_dim0=256 (bits79:48); tensor_dim1=1024
    // (bits111:80); tile_dim0=256; tile_dim1=32; tile_dim2=4;
    // dim0_stride=256; dim1_stride=65536 (gate jump, bits255:208 -> hi dword=1)
    i32x8 g1 = { 0x10000, (int)(256u << 16), (int)(1024u << 16), (int)(256u << 16),
                 (int)(32u | (4u << 16)), 256, 0, 1 };
    i32x4 g2 = { 4, 0, 0, 0 };                           // tensor_dim2=4
    i32x4 g3 = { 0, 0, 0, 0 };
#if __clang_major__ >= 23
    i32x8 g4 = { 0, 0, 0, 0, 0, 0, 0, 0 };
    __builtin_amdgcn_tensor_load_to_lds(g0, g1, g2, g3, g4, 0);
#else
    __builtin_amdgcn_tensor_load_to_lds(g0, g1, g2, g3, 0);
#endif
    __builtin_amdgcn_s_wait_tensorcnt(0);
  }
#else
  for (int i = threadIdx.x; i < (128 * 256) / 8; i += blockDim.x) {
    int rl = i >> 5;
    int ck = (i & 31) * 8;
    int gate = rl >> 5, jj = rl & 31;
    *(f32x4*)(Ws + rl * 256 + ck) =
        *(const f32x4*)(whh + (size_t)(gate * 256 + j0 + jj) * 256 + ck);
  }
#endif
  __syncthreads();

  const int lane = threadIdx.x & 31;
  const int wv   = threadIdx.x >> 5;
  const int half = lane >> 4;
  const int nl   = lane & 15;

  float c[2][8];
  #pragma unroll
  for (int p = 0; p < 2; ++p)
    #pragma unroll
    for (int v = 0; v < 8; ++v) c[p][v] = 0.0f;

  for (int s = 0; s < T; ++s) {
    const int t  = rev ? (T - 1 - s) : s;
    const int tp = rev ? (t + 1) : (t - 1);

    f32x8 acc[8] = {};
    if (s > 0) {
      const __bf16* hrow = hbuf + ((size_t)tp * 128 + wv * 16 + nl) * hw + colbase;
      Frag32 a[8];                                       // all A frags in flight first
      #pragma unroll
      for (int kt = 0; kt < 8; ++kt) {
        a[kt].f4[0] = *(const f32x4*)(hrow + kt * 32 + half * 8);
        a[kt].f4[1] = *(const f32x4*)(hrow + kt * 32 + 16 + half * 8);
      }
      #pragma unroll
      for (int kt = 0; kt < 8; ++kt) {
        #pragma unroll
        for (int nt = 0; nt < 8; ++nt) {
          const __bf16* wr = Ws + (nt * 16 + nl) * 256 + kt * 32 + half * 8;
          Frag32 b;
          b.f4[0] = *(const f32x4*)(wr);
          b.f4[1] = *(const f32x4*)(wr + 16);
          acc[nt] = __builtin_amdgcn_wmma_f32_16x16x32_bf16(
              false, a[kt].v, false, b.v, (short)0, acc[nt], false, false);
        }
      }
    }

    // prefetch next step's pre-activations (global_prefetch_b8)
    if (s + 1 < T) {
      const int tn = rev ? (t - 1) : (t + 1);
      __builtin_prefetch(pre + ((size_t)tn * 128 + wv * 16) * 1024 + j0, 0, 1);
    }

    // gate math: acc[0..1]=i, acc[2..3]=f, acc[4..5]=g, acc[6..7]=o
    #pragma unroll
    for (int p = 0; p < 2; ++p) {
      const int jj = p * 16 + nl;
      #pragma unroll
      for (int v = 0; v < 8; ++v) {
        const int m = wv * 16 + v + 8 * half;            // batch index
        const size_t prow = ((size_t)t * 128 + m) * 1024 + j0 + jj;
        float iv = acc[0 + p][v] + pre[prow];
        float fv = acc[2 + p][v] + pre[prow + 256];
        float gv = acc[4 + p][v] + pre[prow + 512];
        float ov = acc[6 + p][v] + pre[prow + 768];
        float cc = sigm(fv) * c[p][v] + sigm(iv) * tanhf(gv);
        c[p][v] = cc;
        float hh = sigm(ov) * tanhf(cc);
        hbuf[((size_t)t * 128 + m) * hw + colbase + j0 + jj] = (__bf16)hh;
        if (uout) uout[((size_t)t * 128 + m) * 256 + j0 + jj] = hh;
      }
    }

    if (s + 1 < T) {                                     // device-scope barrier
      __syncthreads();
      if (threadIdx.x == 0) {
        __threadfence();
        atomicAdd(cnt, 1u);
        const unsigned target = 8u * (unsigned)(s + 1);
        while (atomicAdd(cnt, 0u) < target) __builtin_amdgcn_s_sleep(2);
      }
      __syncthreads();
      __threadfence();                                   // acquire before reading h
    }
  }
}

// ---- tanh(u) -> bf16 ------------------------------------------------------
__global__ void k_tanh(const float* __restrict__ u, __bf16* __restrict__ d, long n) {
  long i = (long)blockIdx.x * blockDim.x + threadIdx.x;
  if (i < n) d[i] = (__bf16)tanhf(u[i]);
}

// ---- softmax over batch axis + time pooling ------------------------------
__global__ __launch_bounds__(128)
void k_attn(const float* __restrict__ z, const float* __restrict__ attnH,
            float* __restrict__ pooled) {
  const int h = blockIdx.x;
  const int b = threadIdx.x;
  __shared__ float red[128];
  float acc = 0.0f;
  for (int t = 0; t < 512; ++t) {
    float v = z[((size_t)t * 256 + h) * 128 + b];
    red[b] = v; __syncthreads();
    for (int st = 64; st > 0; st >>= 1) {
      if (b < st) red[b] = fmaxf(red[b], red[b + st]);
      __syncthreads();
    }
    float mx = red[0]; __syncthreads();
    float e = __expf(v - mx);
    red[b] = e; __syncthreads();
    for (int st = 64; st > 0; st >>= 1) {
      if (b < st) red[b] += red[b + st];
      __syncthreads();
    }
    float sum = red[0]; __syncthreads();
    acc += e / sum;
  }
  pooled[(size_t)b * 256 + h] = attnH[h] * acc;
}

// ---- final tiny GEMM ------------------------------------------------------
__global__ void k_final(const float* __restrict__ pooled, const float* __restrict__ linW,
                        const float* __restrict__ linb, float* __restrict__ out) {
  int id = blockIdx.x * blockDim.x + threadIdx.x;
  if (id >= 128 * 15) return;
  int b = id / 15, o = id % 15;
  float s = linb[o];
  for (int h = 0; h < 256; ++h) s += pooled[b * 256 + h] * linW[o * 256 + h];
  out[id] = s;
}

// ---------------------------------------------------------------------------
extern "C" void kernel_launch(void* const* d_in, const int* in_sizes, int n_in,
                              void* d_out, int out_size, void* d_ws, size_t ws_size,
                              hipStream_t stream) {
  (void)in_sizes; (void)n_in; (void)out_size; (void)ws_size;
  const float* x       = (const float*)d_in[0];
  const float* wih_l0f = (const float*)d_in[1];
  const float* whh_l0f = (const float*)d_in[2];
  const float* b_l0f   = (const float*)d_in[3];
  const float* wih_l0b = (const float*)d_in[4];
  const float* whh_l0b = (const float*)d_in[5];
  const float* b_l0b   = (const float*)d_in[6];
  const float* wih_l1f = (const float*)d_in[7];
  const float* whh_l1f = (const float*)d_in[8];
  const float* b_l1f   = (const float*)d_in[9];
  const float* wih_l1b = (const float*)d_in[10];
  const float* whh_l1b = (const float*)d_in[11];
  const float* b_l1b   = (const float*)d_in[12];
  const float* wih_u   = (const float*)d_in[13];
  const float* whh_u   = (const float*)d_in[14];
  const float* b_u     = (const float*)d_in[15];
  const float* attn_W  = (const float*)d_in[16];
  const float* attn_b  = (const float*)d_in[17];
  const float* attn_H  = (const float*)d_in[18];
  const float* lin_W   = (const float*)d_in[19];
  const float* lin_b   = (const float*)d_in[20];

  char* ws = (char*)d_ws;
  const size_t MB  = 1024ull * 1024ull;
  const size_t PRE = (size_t)512 * 128 * 1024 * 4;        // 256 MB per pre buffer

  float*  preA   = (float*) (ws);
  float*  preB   = (float*) (ws + PRE);
  __bf16* h0     = (__bf16*)(ws + 2 * PRE);               // (T,B,512) bf16, 64 MB
  __bf16* h1     = (__bf16*)(ws + 2 * PRE + 64 * MB);
  __bf16* hexu   = (__bf16*)(ws + 2 * PRE + 128 * MB);    // top-LSTM h exchange, 32 MB
  float*  u      = (float*) (ws + 2 * PRE + 160 * MB);    // (T,B,256) fp32, 64 MB
  float*  z      = (float*) (ws + 2 * PRE + 224 * MB);    // (T,H,B) fp32, 64 MB
  __bf16* tanhu  = (__bf16*)(ws + 2 * PRE + 288 * MB);    // 32 MB
  float*  pooled = (float*) (ws + 2 * PRE + 320 * MB);    // 128 KB
  char*   wb     = ws + 2 * PRE + 321 * MB;
  __bf16* whh_l0f_b = (__bf16*)(wb + 0 * 512 * 1024);
  __bf16* whh_l0b_b = (__bf16*)(wb + 1 * 512 * 1024);
  __bf16* whh_l1f_b = (__bf16*)(wb + 2 * 512 * 1024);
  __bf16* whh_l1b_b = (__bf16*)(wb + 3 * 512 * 1024);
  __bf16* whh_u_b   = (__bf16*)(wb + 4 * 512 * 1024);
  __bf16* wih_l1f_b = (__bf16*)(wb + 5 * 512 * 1024);
  __bf16* wih_l1b_b = (__bf16*)(wb + 5 * 512 * 1024 + 1 * MB);
  __bf16* wih_u_b   = (__bf16*)(wb + 5 * 512 * 1024 + 2 * MB);
  __bf16* attn_WT   = (__bf16*)(wb + 5 * 512 * 1024 + 3 * MB);
  unsigned* cnts    = (unsigned*)(ws + 2 * PRE + 336 * MB);

  hipMemsetAsync(cnts, 0, 256, stream);                   // barrier counters (capture-safe)

  // weight conversions to bf16
  {
    long n = 1024 * 256;
    int g = (int)((n + 255) / 256);
    k_cvt<<<g, 256, 0, stream>>>(whh_l0f, whh_l0f_b, n);
    k_cvt<<<g, 256, 0, stream>>>(whh_l0b, whh_l0b_b, n);
    k_cvt<<<g, 256, 0, stream>>>(whh_l1f, whh_l1f_b, n);
    k_cvt<<<g, 256, 0, stream>>>(whh_l1b, whh_l1b_b, n);
    k_cvt<<<g, 256, 0, stream>>>(whh_u,   whh_u_b,   n);
  }
  {
    long n = 1024 * 512;
    int g = (int)((n + 255) / 256);
    k_cvt<<<g, 256, 0, stream>>>(wih_l1f, wih_l1f_b, n);
    k_cvt<<<g, 256, 0, stream>>>(wih_l1b, wih_l1b_b, n);
    k_cvt<<<g, 256, 0, stream>>>(wih_u,   wih_u_b,   n);
  }
  k_cvtT<<<(65536 + 255) / 256, 256, 0, stream>>>(attn_W, attn_WT, 256, 256);

  // layer 0: input projection (K=15 scalar), then bidirectional recurrence
  k_proj0<<<262144, 256, 0, stream>>>(x, wih_l0f, b_l0f, wih_l0b, b_l0b, preA, preB);
  k_lstm<<<16, 256, 65536, stream>>>(preA, preB, whh_l0f_b, whh_l0b_b,
                                     h0, 512, nullptr, cnts + 0, 512);

  // layer 1: WMMA input projections + bidirectional recurrence
  dim3 gg(1024 / 64, 65536 / 128);
  k_gemm<<<gg, 256, 0, stream>>>(h0, wih_l1f_b, b_l1f, preA, 1024, 512, 0);
  k_gemm<<<gg, 256, 0, stream>>>(h0, wih_l1b_b, b_l1b, preB, 1024, 512, 0);
  k_lstm<<<16, 256, 65536, stream>>>(preA, preB, whh_l1f_b, whh_l1b_b,
                                     h1, 512, nullptr, cnts + 2, 512);

  // top LSTM (unidirectional)
  k_gemm<<<gg, 256, 0, stream>>>(h1, wih_u_b, b_u, preA, 1024, 512, 0);
  k_lstm<<<8, 256, 65536, stream>>>(preA, nullptr, whh_u_b, nullptr,
                                    hexu, 256, u, cnts + 4, 512);

  // attention: tanh -> WMMA GEMM (transposed store to (T,H,B)) -> softmax/pool
  long nu = (long)512 * 128 * 256;
  k_tanh<<<(int)((nu + 255) / 256), 256, 0, stream>>>(u, tanhu, nu);
  dim3 gz(256 / 64, 65536 / 128);
  k_gemm<<<gz, 256, 0, stream>>>(tanhu, attn_WT, attn_b, z, 256, 256, 1);
  k_attn<<<256, 128, 0, stream>>>(z, attn_H, pooled);
  k_final<<<8, 256, 0, stream>>>(pooled, lin_W, lin_b, (float*)d_out);
}
